// SequentialAudioVisualModel_77627238908591
// MI455X (gfx1250) — matrix operand order repaired and checked by reading, not compiled, not run on hardware
//
#include <hip/hip_runtime.h>
#include <hip/hip_bf16.h>
#include <math.h>
#include <stdint.h>

// ---------------------------------------------------------------------------
// MI455X (gfx1250) implementation.
// Roofline: output is 40960x4096 f32 = 671MB -> ~29us store floor @ 23.3TB/s.
// Classifier GEMM = 42.9 GFLOP in f16 WMMA (f32 acc) -> far below store floor.
// h2 activations (10.5MB f16) + Wclf (1MB f16) stay L2-resident (192MB L2).
// Wclf/Wf2 slices are staged into LDS per block via the CDNA5 async
// global->LDS copy path (ASYNCcnt + s_wait_asynccnt), so every wave reads
// B fragments from LDS (ds_load_b128) instead of 8x re-reading L2.
// Final stores use the NT temporal hint to keep the streamed output from
// displacing the L2-resident working set.
// ---------------------------------------------------------------------------

typedef __attribute__((ext_vector_type(16))) _Float16 v16h;
typedef __attribute__((ext_vector_type(8)))  _Float16 v8h;
typedef __attribute__((ext_vector_type(8)))  float    v8f;

#define D_    128
#define DINO_ 384
#define V_    4096
#define H_    4
#define T_    40
#define P_    256
#define B_    4

// Async copy of one 16-byte chunk from global to LDS (CDNA5, ASYNCcnt).
// Generic __shared__ pointers carry the LDS byte address in their low 32 bits
// (flat->LDS mapping truncates to addr[31:0]), so we pass that as VDST.
__device__ __forceinline__ void async_copy_b128(const void* gsrc, void* lds_dst) {
  uint32_t lds_addr = (uint32_t)(uintptr_t)lds_dst;
  asm volatile("global_load_async_to_lds_b128 %0, %1, off"
               :: "v"(lds_addr), "v"(gsrc) : "memory");
}
__device__ __forceinline__ void async_wait_all() {
  asm volatile("s_wait_asynccnt 0x0" ::: "memory");
}

// ---------------------------------------------------------------------------
// Kernel 1: tiny causal transformer encoder, one workgroup per batch item.
// All scratch in global ws; __syncthreads orders phases. Negligible cost.
// ---------------------------------------------------------------------------
__global__ __launch_bounds__(256)
void encoder_kernel(const int* __restrict__ tokens, const float* __restrict__ emb,
                    const float* __restrict__ pos, const float* __restrict__ Wqkv,
                    const float* __restrict__ bqkv, const float* __restrict__ Wo,
                    const float* __restrict__ bo, const float* __restrict__ ln1g,
                    const float* __restrict__ ln1b, const float* __restrict__ W1,
                    const float* __restrict__ b1, const float* __restrict__ W2,
                    const float* __restrict__ b2, const float* __restrict__ ln2g,
                    const float* __restrict__ ln2b,
                    float* xbuf, float* qkv, float* ao, float* x1, float* hr,
                    float* ctx)
{
  const int b = blockIdx.x;
  const int tid = threadIdx.x;
  xbuf += b * T_ * D_;
  qkv  += b * T_ * 3 * D_;
  ao   += b * T_ * D_;
  x1   += b * T_ * D_;
  hr   += b * T_ * D_;
  ctx  += b * T_ * D_;
  __shared__ float s_m[T_], s_v[T_];

  // x = emb[shift_right(tokens)] + pos
  for (int idx = tid; idx < T_ * D_; idx += 256) {
    int t = idx / D_, d = idx % D_;
    int tok = (t == 0) ? 0 : tokens[b * T_ + t - 1];
    xbuf[idx] = emb[tok * D_ + d] + pos[idx];
  }
  __syncthreads();

  // qkv = x @ Wqkv + bqkv
  for (int idx = tid; idx < T_ * 3 * D_; idx += 256) {
    int t = idx / (3 * D_), c = idx % (3 * D_);
    float acc = bqkv[c];
    for (int k = 0; k < D_; ++k) acc += xbuf[t * D_ + k] * Wqkv[k * 3 * D_ + c];
    qkv[idx] = acc;
  }
  __syncthreads();

  // causal attention: one thread per (head, query)
  if (tid < H_ * T_) {
    int h = tid / T_, q = tid % T_;
    const int dh = D_ / H_;  // 32
    float sc[T_];
    float mx = -1e30f;
    for (int k = 0; k <= q; ++k) {
      float s = 0.f;
      for (int j = 0; j < dh; ++j)
        s += qkv[q * 3 * D_ + h * dh + j] * qkv[k * 3 * D_ + D_ + h * dh + j];
      s *= 0.17677669529663687f;  // 1/sqrt(32)
      sc[k] = s;
      mx = fmaxf(mx, s);
    }
    float sum = 0.f;
    for (int k = 0; k <= q; ++k) { sc[k] = expf(sc[k] - mx); sum += sc[k]; }
    float inv = 1.f / sum;
    for (int j = 0; j < dh; ++j) {
      float o = 0.f;
      for (int k = 0; k <= q; ++k)
        o += sc[k] * qkv[k * 3 * D_ + 2 * D_ + h * dh + j];
      ao[q * D_ + h * dh + j] = o * inv;
    }
  }
  __syncthreads();

  // x1 = x + ao @ Wo + bo
  for (int idx = tid; idx < T_ * D_; idx += 256) {
    int t = idx / D_, d = idx % D_;
    float acc = bo[d];
    for (int k = 0; k < D_; ++k) acc += ao[t * D_ + k] * Wo[k * D_ + d];
    x1[idx] = xbuf[idx] + acc;
  }
  __syncthreads();
  // LayerNorm 1
  if (tid < T_) {
    float m = 0.f;
    for (int k = 0; k < D_; ++k) m += x1[tid * D_ + k];
    m /= (float)D_;
    float v = 0.f;
    for (int k = 0; k < D_; ++k) { float dd = x1[tid * D_ + k] - m; v += dd * dd; }
    s_m[tid] = m;
    s_v[tid] = v / (float)D_;
  }
  __syncthreads();
  for (int idx = tid; idx < T_ * D_; idx += 256) {
    int t = idx / D_, d = idx % D_;
    x1[idx] = (x1[idx] - s_m[t]) * rsqrtf(s_v[t] + 1e-5f) * ln1g[d] + ln1b[d];
  }
  __syncthreads();

  // hr = relu(x1 @ W1 + b1)
  for (int idx = tid; idx < T_ * D_; idx += 256) {
    int t = idx / D_, d = idx % D_;
    float acc = b1[d];
    for (int k = 0; k < D_; ++k) acc += x1[t * D_ + k] * W1[k * D_ + d];
    hr[idx] = fmaxf(acc, 0.f);
  }
  __syncthreads();

  // r = x1 + hr @ W2 + b2 (into xbuf, free now)
  for (int idx = tid; idx < T_ * D_; idx += 256) {
    int t = idx / D_, d = idx % D_;
    float acc = b2[d];
    for (int k = 0; k < D_; ++k) acc += hr[t * D_ + k] * W2[k * D_ + d];
    xbuf[idx] = x1[idx] + acc;
  }
  __syncthreads();
  // LayerNorm 2 -> ctx
  if (tid < T_) {
    float m = 0.f;
    for (int k = 0; k < D_; ++k) m += xbuf[tid * D_ + k];
    m /= (float)D_;
    float v = 0.f;
    for (int k = 0; k < D_; ++k) { float dd = xbuf[tid * D_ + k] - m; v += dd * dd; }
    s_m[tid] = m;
    s_v[tid] = v / (float)D_;
  }
  __syncthreads();
  for (int idx = tid; idx < T_ * D_; idx += 256) {
    int t = idx / D_, d = idx % D_;
    ctx[idx] = (xbuf[idx] - s_m[t]) * rsqrtf(s_v[t] + 1e-5f) * ln2g[d] + ln2b[d];
  }
}

// ---------------------------------------------------------------------------
// Kernel 2: pe = patches @ Wf1[:DINO], ce = ctx @ Wf1[DINO:]
// ---------------------------------------------------------------------------
__global__ __launch_bounds__(128)
void pece_kernel(const float* __restrict__ patches, const float* __restrict__ ctx,
                 const float* __restrict__ Wf1, float* __restrict__ pe,
                 float* __restrict__ ce)
{
  const int r = blockIdx.x;
  const int d = threadIdx.x;
  if (r < B_ * P_) {
    const float* row = patches + (size_t)r * DINO_;
    float acc = 0.f;
    for (int k = 0; k < DINO_; ++k) acc += row[k] * Wf1[k * D_ + d];
    pe[r * D_ + d] = acc;
  } else {
    const int rr = r - B_ * P_;
    const float* row = ctx + rr * D_;
    float acc = 0.f;
    for (int k = 0; k < D_; ++k) acc += row[k] * Wf1[(DINO_ + k) * D_ + d];
    ce[rr * D_ + d] = acc;
  }
}

// ---------------------------------------------------------------------------
// Kernel 3: pack Wf2 and Wclf as f16 in WMMA B-fragment order:
//   fragment(jtile, kchunk): lane -> col n = jtile*16 + (lane&15),
//   element i -> K = 32*kchunk + (i&7) + 16*(i>>3) + 8*(lane>>4).
// Packed index = ((jtile*4 + kchunk)*32 + lane)*16 + i  -> contiguous v16h.
// A contiguous run of jtiles is a contiguous slice -> trivial LDS staging.
// ---------------------------------------------------------------------------
__global__ __launch_bounds__(256)
void pack_kernel(const float* __restrict__ Wf2, const float* __restrict__ Wclf,
                 _Float16* __restrict__ Wf2h, _Float16* __restrict__ Wclfh)
{
  const int idx = blockIdx.x * 256 + threadIdx.x;
  const int nf2 = 8 * 4 * 32 * 16;     // 16384
  const int nclf = 256 * 4 * 32 * 16;  // 524288
  if (idx < nf2) {
    int i = idx & 15, lane = (idx >> 4) & 31, c = (idx >> 9) & 3, j = idx >> 11;
    int n = j * 16 + (lane & 15);
    int k = 32 * c + (i & 7) + 16 * (i >> 3) + 8 * (lane >> 4);
    Wf2h[idx] = (_Float16)Wf2[k * D_ + n];
  } else if (idx < nf2 + nclf) {
    int o = idx - nf2;
    int i = o & 15, lane = (o >> 4) & 31, c = (o >> 9) & 3, j = o >> 11;
    int n = j * 16 + (lane & 15);
    int k = 32 * c + (i & 7) + 16 * (i >> 3) + 8 * (lane >> 4);
    Wclfh[o] = (_Float16)Wclf[k * V_ + n];
  }
}

// ---------------------------------------------------------------------------
// Kernel 4: h2 = gelu(pe[b,p]+ce[b,t]+bf1) @ Wf2 + bf2, stored f16 row-major.
// 8 waves/block; wave handles 16 rows. Wf2 (32KB packed f16) is staged into
// LDS once per block via async global->LDS; all waves read it via ds_load.
// A fragments (16x32 f16): lane holds row m=lane&15; element i ->
//   K = 32*kchunk + (i&7) + 16*(i>>3) + 8*(lane>>4).
// ---------------------------------------------------------------------------
__global__ __launch_bounds__(256)
void h2_kernel(const float* __restrict__ pe, const float* __restrict__ ce,
               const float* __restrict__ bf1, const float* __restrict__ bf2,
               const _Float16* __restrict__ Wf2h, _Float16* __restrict__ h2)
{
  __shared__ _Float16 sW[8 * 4 * 32 * 16];  // 16384 halves = 32KB
  const int tid = threadIdx.x;
  const int wave = tid >> 5, lane = tid & 31;
  const int g = lane >> 4, ml = lane & 15;

  // stage packed Wf2 into LDS: 2048 x b128 async transfers
#pragma unroll
  for (int k = 0; k < 8; ++k) {
    int idx = tid + k * 256;
    async_copy_b128(Wf2h + idx * 8, sW + idx * 8);
  }
  async_wait_all();
  __syncthreads();

  int blk = blockIdx.x;
  const int pblk = blk & 1;  // P_/128 == 2
  blk >>= 1;
  const int t = blk % T_;
  const int b = blk / T_;
  const int p = pblk * 128 + wave * 16 + ml;
  const float* perow = pe + (size_t)(b * P_ + p) * D_;
  const float* cerow = ce + (size_t)(b * T_ + t) * D_;

  v16h a[4];
#pragma unroll
  for (int c = 0; c < 4; ++c) {
#pragma unroll
    for (int i = 0; i < 16; ++i) {
      int K = 32 * c + (i & 7) + ((i >> 3) << 4) + (g << 3);
      float x = perow[K] + cerow[K] + bf1[K];
      float ge = 0.5f * x * (1.f + erff(x * 0.70710678118654752f));  // exact gelu
      a[c][i] = (_Float16)ge;
    }
  }

  const size_t rowbase = (size_t)(b * T_ + t) * P_ + (size_t)(pblk * 128 + wave * 16);
#pragma unroll
  for (int j = 0; j < 8; ++j) {
    float bb = bf2[j * 16 + ml];
    v8f acc = {bb, bb, bb, bb, bb, bb, bb, bb};
#pragma unroll
    for (int c = 0; c < 4; ++c) {
      v16h bfrag = *(const v16h*)(sW + (((j * 4 + c) * 32 + lane) << 4));
      acc = __builtin_amdgcn_wmma_f32_16x16x32_f16(
          false, a[c], false, bfrag, (short)0, acc, false, false);
    }
#pragma unroll
    for (int r = 0; r < 8; ++r) {
      int m = r + 8 * g;  // C/D layout: lanes 0-15 -> M=r, lanes 16-31 -> M=r+8
      h2[(rowbase + m) * D_ + j * 16 + ml] = (_Float16)acc[r];
    }
  }
}

// ---------------------------------------------------------------------------
// Kernel 5: logits = h2 @ Wclf + bclf.  M=40960, N=4096, K=128.
// grid = (M/128, N/256); 8 waves/block (wave = 16-row sub-tile).
// The block's 256-col Wclf slice (64KB packed f16, contiguous) is staged into
// LDS once via async global->LDS, shared by all 8 waves -> 8x less L2 read.
// Each wave: 16 n-tiles x 4 k-chunks = 64 WMMAs; A fragments loaded as two
// contiguous b128 per chunk; f32 results stored with NT hint (write-once
// 671MB stream must not displace L2-resident h2/Wclf).
// ---------------------------------------------------------------------------
__global__ __launch_bounds__(256)
void clf_kernel(const _Float16* __restrict__ h2, const _Float16* __restrict__ Wclfh,
                const float* __restrict__ bclf, float* __restrict__ out)
{
  __shared__ _Float16 sB[16 * 4 * 32 * 16];  // 32768 halves = 64KB
  const int tid = threadIdx.x;
  const int wave = tid >> 5, lane = tid & 31;
  const int g = lane >> 4, ml = lane & 15;

  const int nbase0 = blockIdx.y * 256;        // this block's first column
  const int jj0 = nbase0 >> 4;                // first global n-tile (16 tiles)
  // stage this block's Wclf slice: contiguous 32768 halves from jj0*2048
  const _Float16* src = Wclfh + (size_t)jj0 * 2048;
#pragma unroll
  for (int k = 0; k < 16; ++k) {
    int idx = tid + k * 256;
    async_copy_b128(src + idx * 8, sB + idx * 8);
  }
  async_wait_all();
  __syncthreads();

  const size_t mbase = (size_t)blockIdx.x * 128 + (size_t)wave * 16;
  const _Float16* arow = h2 + (mbase + ml) * D_;

  // A fragments: per chunk c, lane's 16 halves are two contiguous runs of 8
  // at K = 32c + 8g and K = 32c + 8g + 16 -> two b128 loads.
  v16h a[4];
#pragma unroll
  for (int c = 0; c < 4; ++c) {
    v8h lo = *(const v8h*)(arow + 32 * c + 8 * g);
    v8h hi = *(const v8h*)(arow + 32 * c + 8 * g + 16);
#pragma unroll
    for (int i = 0; i < 8; ++i) { a[c][i] = lo[i]; a[c][i + 8] = hi[i]; }
  }

#pragma unroll
  for (int j = 0; j < 16; ++j) {
    const int n0 = nbase0 + j * 16;
    float bb = bclf[n0 + ml];
    v8f acc = {bb, bb, bb, bb, bb, bb, bb, bb};
#pragma unroll
    for (int c = 0; c < 4; ++c) {
      v16h bfrag = *(const v16h*)(sB + (((j * 4 + c) * 32 + lane) << 4));
      acc = __builtin_amdgcn_wmma_f32_16x16x32_f16(
          false, a[c], false, bfrag, (short)0, acc, false, false);
    }
#pragma unroll
    for (int r = 0; r < 8; ++r) {
      __builtin_nontemporal_store(
          acc[r], &out[(mbase + (size_t)(r + 8 * g)) * V_ + n0 + ml]);
    }
  }
}

// ---------------------------------------------------------------------------
extern "C" void kernel_launch(void* const* d_in, const int* in_sizes, int n_in,
                              void* d_out, int out_size, void* d_ws, size_t ws_size,
                              hipStream_t stream) {
  (void)in_sizes; (void)n_in; (void)out_size; (void)ws_size;
  const float* patches = (const float*)d_in[0];
  const int*   tokens  = (const int*)d_in[1];
  const float* emb     = (const float*)d_in[2];
  const float* pos     = (const float*)d_in[3];
  const float* Wqkv    = (const float*)d_in[4];
  const float* bqkv    = (const float*)d_in[5];
  const float* Wo      = (const float*)d_in[6];
  const float* bo      = (const float*)d_in[7];
  const float* ln1g    = (const float*)d_in[8];
  const float* ln1b    = (const float*)d_in[9];
  const float* W1      = (const float*)d_in[10];
  const float* b1      = (const float*)d_in[11];
  const float* W2      = (const float*)d_in[12];
  const float* b2      = (const float*)d_in[13];
  const float* ln2g    = (const float*)d_in[14];
  const float* ln2b    = (const float*)d_in[15];
  const float* Wf1     = (const float*)d_in[16];
  const float* bf1     = (const float*)d_in[17];
  const float* Wf2     = (const float*)d_in[18];
  const float* bf2     = (const float*)d_in[19];
  const float* Wclf    = (const float*)d_in[20];
  const float* bclf    = (const float*)d_in[21];
  float* out = (float*)d_out;

  // workspace carve-up (256B aligned slices), total ~12.3 MB
  char* w = (char*)d_ws;
  auto take = [&](size_t bytes) {
    char* p = w;
    w += (bytes + 255) & ~(size_t)255;
    return p;
  };
  float* ctx  = (float*)take(sizeof(float) * B_ * T_ * D_);
  float* pe   = (float*)take(sizeof(float) * B_ * P_ * D_);
  float* ce   = (float*)take(sizeof(float) * B_ * T_ * D_);
  float* xbuf = (float*)take(sizeof(float) * B_ * T_ * D_);
  float* qkv  = (float*)take(sizeof(float) * B_ * T_ * 3 * D_);
  float* ao   = (float*)take(sizeof(float) * B_ * T_ * D_);
  float* x1   = (float*)take(sizeof(float) * B_ * T_ * D_);
  float* hr   = (float*)take(sizeof(float) * B_ * T_ * D_);
  _Float16* Wf2h  = (_Float16*)take(sizeof(_Float16) * D_ * D_);
  _Float16* Wclfh = (_Float16*)take(sizeof(_Float16) * D_ * V_);
  _Float16* h2    = (_Float16*)take(sizeof(_Float16) * (size_t)B_ * T_ * P_ * D_);

  // 1) token encoder (tiny)
  encoder_kernel<<<B_, 256, 0, stream>>>(tokens, emb, pos, Wqkv, bqkv, Wo, bo,
                                         ln1g, ln1b, W1, b1, W2, b2, ln2g, ln2b,
                                         xbuf, qkv, ao, x1, hr, ctx);
  // 2) pe / ce projections
  pece_kernel<<<B_ * P_ + B_ * T_, 128, 0, stream>>>(patches, ctx, Wf1, pe, ce);
  // 3) weight packing to f16 WMMA B-fragments (independent of 1-2)
  pack_kernel<<<(16384 + 524288) / 256, 256, 0, stream>>>(Wf2, Wclf, Wf2h, Wclfh);
  // 4) h2 = gelu(pe+ce+bf1) @ Wf2 + bf2  (f16, 10.5MB -> L2 resident)
  h2_kernel<<<B_ * T_ * (P_ / 128), 256, 0, stream>>>(pe, ce, bf1, bf2, Wf2h, h2);
  // 5) logits = h2 @ Wclf + bclf  (store-bandwidth bound)
  dim3 cgrid((B_ * T_ * P_) / 128, V_ / 256);
  clf_kernel<<<cgrid, 256, 0, stream>>>(h2, Wclfh, bclf, out);
}
